// CantorAttention_51213190038176
// MI455X (gfx1250) — matrix operand order repaired and checked by reading, not compiled, not run on hardware
//
#include <hip/hip_runtime.h>
#include <hip/hip_bf16.h>

// ---------------------------------------------------------------------------
// Problem constants (match reference)
// ---------------------------------------------------------------------------
#define BB   2
#define SS   4096
#define DIM  1024
#define HH   16
#define HD   64
#define KN   64
#define SCALE 0.125f            // 1/sqrt(64)

typedef __attribute__((ext_vector_type(16))) __bf16    v16bf;
typedef __attribute__((ext_vector_type(8)))  float     v8f;
typedef __attribute__((ext_vector_type(8)))  unsigned  v8u;

// Pack two fp32 into one dword of two bf16 (lo = first).  Lowers to
// v_cvt_pk_bf16_f32.
static __device__ __forceinline__ unsigned pack_bf16(float lo, float hi) {
    unsigned short a = __builtin_bit_cast(unsigned short, (__bf16)lo);
    unsigned short b = __builtin_bit_cast(unsigned short, (__bf16)hi);
    return (unsigned)a | ((unsigned)b << 16);
}

// Build a 16-bf16 A/B fragment from 4 float4 (two contiguous 8-float runs).
static __device__ __forceinline__ v16bf frag_from_f32(float4 a0, float4 a1,
                                                      float4 a2, float4 a3) {
    v8u u;
    u[0] = pack_bf16(a0.x, a0.y);
    u[1] = pack_bf16(a0.z, a0.w);
    u[2] = pack_bf16(a1.x, a1.y);
    u[3] = pack_bf16(a1.z, a1.w);
    u[4] = pack_bf16(a2.x, a2.y);
    u[5] = pack_bf16(a2.z, a2.w);
    u[6] = pack_bf16(a3.x, a3.y);
    u[7] = pack_bf16(a3.z, a3.w);
    return __builtin_bit_cast(v16bf, u);
}

// ---------------------------------------------------------------------------
// C[M,N] = A[M,K] @ B[K,N] + bias[N]   (fp32 in/out, bf16 WMMA accumulate f32)
//
// Workgroup: 256 threads = 8 waves; tile 256(M) x 64(N); K stepped by 32.
// Wave w owns rows [m0+32w, m0+32w+32): 2 A-fragments x 4 B-fragments
// = 8 WMMAs per K-step against 6 fragment loads (B reused 2x from LDS).
//
// B tile (32x64) staged in LDS as packed bf16 pairs, transposed to
// sB[n][kpair], row stride 20 dwords (80 B -> rows stay 16 B aligned).
//   stage:   thread (kpair, n4) loads float4 from rows k+2*kpair, k+2*kpair+1
//            -> 4 packed dwords -> sB[n4+j][kpair]
//   consume: lane fragment = 4 dwords at row n = f*16+(lane&15), col (lane>>4)*4,
//            plus 4 dwords at col 8+(lane>>4)*4  -> two ds_load_b128.
// A fragments loaded direct from global (ISA 16-bit A layout: lane L:
// M = L&15, K runs khalf..khalf+7 and 16+khalf..+7, khalf = (L>>4)*8).
// C/D layout: lane L: N = L&15, acc elem r -> M = (L>>4)*8 + r.
// ---------------------------------------------------------------------------
__global__ __launch_bounds__(256)
void gemm_bf16_wmma(const float* __restrict__ A,
                    const float* __restrict__ Bm,
                    const float* __restrict__ bias,
                    float* __restrict__ C,
                    int M, int N, int K)
{
    __shared__ __align__(16) unsigned sB[64 * 20];   // [n][kpair], stride 20

    const int tid  = threadIdx.x;
    const int lane = tid & 31;
    const int wave = tid >> 5;

    const int m0 = blockIdx.x * 256 + wave * 32;     // 32-row strip per wave
    const int n0 = blockIdx.y * 64;                  // 64-col strip (4 frags)

    const int col   = lane & 15;
    const int half  = lane >> 4;
    const int khalf = half * 8;

    // B staging coordinates for this thread
    const int bkp = tid >> 4;                        // kpair 0..15
    const int bn4 = (tid & 15) * 4;                  // n offset 0,4,...,60

    v8f acc[2][4] = {};

    const float* arow0 = A + (size_t)(m0 + col) * K;        // A frag 0 row
    const float* arow1 = A + (size_t)(m0 + 16 + col) * K;   // A frag 1 row

    for (int k = 0; k < K; k += 32) {
        // ---- cooperative B stage: 32x64 fp32 -> packed bf16 in LDS ----
        __syncthreads();
        {
            const float* b0 = Bm + (size_t)(k + 2 * bkp) * N + n0 + bn4;
            const float* b1 = b0 + N;
            float4 g0 = *reinterpret_cast<const float4*>(b0);
            float4 g1 = *reinterpret_cast<const float4*>(b1);
            sB[(bn4 + 0) * 20 + bkp] = pack_bf16(g0.x, g1.x);
            sB[(bn4 + 1) * 20 + bkp] = pack_bf16(g0.y, g1.y);
            sB[(bn4 + 2) * 20 + bkp] = pack_bf16(g0.z, g1.z);
            sB[(bn4 + 3) * 20 + bkp] = pack_bf16(g0.w, g1.w);
        }
        __syncthreads();

        // ---- A fragments: two contiguous 8-float runs each ----
        const float4* p00 = reinterpret_cast<const float4*>(arow0 + k + khalf);
        const float4* p01 = reinterpret_cast<const float4*>(arow0 + k + 16 + khalf);
        v16bf av0 = frag_from_f32(p00[0], p00[1], p01[0], p01[1]);

        const float4* p10 = reinterpret_cast<const float4*>(arow1 + k + khalf);
        const float4* p11 = reinterpret_cast<const float4*>(arow1 + k + 16 + khalf);
        v16bf av1 = frag_from_f32(p10[0], p10[1], p11[0], p11[1]);

        // hint the next K-step's A lines into cache (global_prefetch_b8)
        if (k + 32 < K) {
            __builtin_prefetch(arow0 + k + 32 + khalf, 0, 0);
            __builtin_prefetch(arow1 + k + 32 + khalf, 0, 0);
        }

        // ---- 8 WMMAs: each B fragment (2x ds_load_b128) reused twice ----
#pragma unroll
        for (int f = 0; f < 4; ++f) {
            const unsigned* p = &sB[(f * 16 + col) * 20 + half * 4];
            uint4 u0 = *reinterpret_cast<const uint4*>(p);
            uint4 u1 = *reinterpret_cast<const uint4*>(p + 8);
            v8u bu;
            bu[0] = u0.x; bu[1] = u0.y; bu[2] = u0.z; bu[3] = u0.w;
            bu[4] = u1.x; bu[5] = u1.y; bu[6] = u1.z; bu[7] = u1.w;
            v16bf bv = __builtin_bit_cast(v16bf, bu);

            acc[0][f] = __builtin_amdgcn_wmma_f32_16x16x32_bf16(
                            false, av0, false, bv, (short)0, acc[0][f],
                            false, false);
            acc[1][f] = __builtin_amdgcn_wmma_f32_16x16x32_bf16(
                            false, av1, false, bv, (short)0, acc[1][f],
                            false, false);
        }
    }

    // ---- epilogue: bias add, fp32 store per C/D layout ----
#pragma unroll
    for (int i = 0; i < 2; ++i) {
#pragma unroll
        for (int f = 0; f < 4; ++f) {
            const int n  = n0 + f * 16 + col;
            const float bn = bias[n];
#pragma unroll
            for (int r = 0; r < 8; ++r) {
                const int mr = m0 + i * 16 + half * 8 + r;
                C[(size_t)mr * N + n] = acc[i][f][r] + bn;
            }
        }
    }
}

// ---------------------------------------------------------------------------
// Routed attention.  One 64-thread block (2 waves) per (b,h,q).
// qkv layout: [B*S, 3*DIM]; q at +h*64, k at +DIM+h*64, v at +2*DIM+h*64.
// Phase 1: thread t owns route t  -> score_t = SCALE * dot(q, k[route_t]).
// Phase 2: thread t owns dim   t  -> out_t   = sum_kn p_kn * v[route_kn][t].
// Gathers hit L2 (whole k/v tensor = 64 MB << 192 MB L2).
// ---------------------------------------------------------------------------
__global__ __launch_bounds__(64)
void cantor_attn(const float* __restrict__ qkv,
                 const int*   __restrict__ routes,
                 float* __restrict__ out)
{
    const int idx = blockIdx.x;
    const int q = idx % SS;
    const int h = (idx / SS) % HH;
    const int b = idx / (SS * HH);
    const int t = threadIdx.x;

    __shared__ float sh_q[HD];
    __shared__ float sh_p[KN];
    __shared__ int   sh_r[KN];
    __shared__ float sh_s[KN];

    const size_t qbase = (size_t)(b * SS + q) * (3 * DIM);
    sh_q[t] = qkv[qbase + h * HD + t];
    sh_r[t] = routes[q * KN + t];
    __syncthreads();

    // ---- phase 1: scores ----
    const int r = sh_r[t];
    const float* krow = qkv + (size_t)(b * SS + r) * (3 * DIM) + DIM + h * HD;
    float s = 0.0f;
#pragma unroll
    for (int d4 = 0; d4 < HD / 4; ++d4) {
        float4 kv = reinterpret_cast<const float4*>(krow)[d4];
        s += sh_q[4 * d4 + 0] * kv.x;
        s += sh_q[4 * d4 + 1] * kv.y;
        s += sh_q[4 * d4 + 2] * kv.z;
        s += sh_q[4 * d4 + 3] * kv.w;
    }
    s *= SCALE;

    // ---- softmax over 64 routes (LDS, redundant per-thread reduce) ----
    sh_s[t] = s;
    __syncthreads();
    float mx = -3.402823466e+38f;
#pragma unroll 8
    for (int i = 0; i < KN; ++i) mx = fmaxf(mx, sh_s[i]);
    const float p = __expf(s - mx);
    sh_p[t] = p;
    __syncthreads();
    float den = 0.0f;
#pragma unroll 8
    for (int i = 0; i < KN; ++i) den += sh_p[i];
    const float inv = 1.0f / den;

    // ---- phase 2: out[d] = sum_kn p_kn * v[route_kn][d]  (d = t) ----
    float o = 0.0f;
#pragma unroll 4
    for (int kn = 0; kn < KN; ++kn) {
        const int rk = sh_r[kn];
        const float* vrow =
            qkv + (size_t)(b * SS + rk) * (3 * DIM) + 2 * DIM + h * HD;
        o += sh_p[kn] * vrow[t];          // consecutive t -> coalesced
    }
    out[(size_t)(b * SS + q) * DIM + h * HD + t] = o * inv;
}

// ---------------------------------------------------------------------------
// Launcher: qkv-GEMM -> routed attention -> out-GEMM
// ---------------------------------------------------------------------------
extern "C" void kernel_launch(void* const* d_in, const int* in_sizes, int n_in,
                              void* d_out, int out_size, void* d_ws, size_t ws_size,
                              hipStream_t stream)
{
    const float* x      = (const float*)d_in[0];   // [B,S,DIM]
    const float* W_qkv  = (const float*)d_in[1];   // [DIM, 3*DIM]
    const float* b_qkv  = (const float*)d_in[2];   // [3*DIM]
    const float* W_out  = (const float*)d_in[3];   // [DIM, DIM]
    const float* b_out  = (const float*)d_in[4];   // [DIM]
    const int*   routes = (const int*)  d_in[5];   // [S, KN]

    float* out = (float*)d_out;

    const int Mrows = BB * SS;                     // 8192
    float* qkv  = (float*)d_ws;                                      // 96 MB
    float* attn = (float*)((char*)d_ws +
                           (size_t)Mrows * 3 * DIM * sizeof(float)); // +32 MB

    // 1) qkv = x @ W_qkv + b_qkv        (M=8192, N=3072, K=1024)
    {
        dim3 grid(Mrows / 256, (3 * DIM) / 64);
        gemm_bf16_wmma<<<grid, 256, 0, stream>>>(x, W_qkv, b_qkv, qkv,
                                                 Mrows, 3 * DIM, DIM);
    }

    // 2) routed attention               (one block per (b,h,q))
    {
        dim3 grid(BB * HH * SS);
        cantor_attn<<<grid, 64, 0, stream>>>(qkv, routes, attn);
    }

    // 3) out = attn @ W_out + b_out     (M=8192, N=1024, K=1024)
    {
        dim3 grid(Mrows / 256, DIM / 64);
        gemm_bf16_wmma<<<grid, 256, 0, stream>>>(attn, W_out, b_out, out,
                                                 Mrows, DIM, DIM);
    }
}
